// WaveNet_16114717295015
// MI455X (gfx1250) — compile-verified, matrix operand order checked by reading, not run on hardware
//
#include <hip/hip_runtime.h>

typedef __attribute__((ext_vector_type(16))) _Float16 v16h;
typedef __attribute__((ext_vector_type(8)))  _Float16 v8h;
typedef __attribute__((ext_vector_type(8)))  float    v8f;
typedef __attribute__((ext_vector_type(4)))  float    v4f;
typedef __attribute__((ext_vector_type(4)))  unsigned int u32x4;

#define TILE   512
#define HALO   1152
#define HROWS  (TILE + HALO)      /* 1664 */
#define NL     9
#define TTOT   32768
#define NB     8

#define NFRAG_CONV  (NL*8*3*2)               /* 432 */
#define FRAG_SKIP0  NFRAG_CONV               /* 432 */
#define FRAG_OUT0   (FRAG_SKIP0 + NL*4*2)    /* 504 */
#define FRAG_OUT10  (FRAG_OUT0 + NL*4*2)     /* 576 */
#define NFRAGS      (FRAG_OUT10 + 8)         /* 584 */

#define WMMA_F16(A,B,C) \
  __builtin_amdgcn_wmma_f32_16x16x32_f16(false,(A),false,(B),(short)0,(C),false,false)

// one v_exp_f32 + v_rcp_f32 per activation
__device__ __forceinline__ float fast_sigmoid(float x) {
  return __builtin_amdgcn_rcpf(1.f + __expf(-x));
}
__device__ __forceinline__ float fast_tanh(float x) {
  return 2.f * fast_sigmoid(2.f * x) - 1.f;
}

// A-fragment: packed by wavenet_pack as [frag][lane][16 halves] -> one 32B load/lane
__device__ __forceinline__ v16h load_afrag(const _Float16* __restrict__ pack, int frag, int lane) {
  const _Float16* p = pack + ((size_t)frag << 9) + (lane << 4);
  return *(const v16h*)p;
}

// B-fragment (K=channel, N=time) from a [row][64ch] f16 LDS buffer.
// K pattern per lane: halves 0..7 -> K = kf*32 + hi*8 + 0..7 ; halves 8..15 -> +16.
__device__ __forceinline__ v16h load_bfrag(const _Float16* base, int row, int kf, int hi) {
  const _Float16* p = base + row * 64 + kf * 32 + hi * 8;
  v8h lo = *(const v8h*)p;
  v8h hv = *(const v8h*)(p + 16);
  return __builtin_shufflevector(lo, hv, 0,1,2,3,4,5,6,7,8,9,10,11,12,13,14,15);
}

// Bias in C/D layout: VGPR r, lane l -> channel m*16 + r + 8*(l>=16)
__device__ __forceinline__ v8f load_biasC(const float* bf, int m, int hi) {
  const float* p = bf + m * 16 + hi * 8;
  v4f a = *(const v4f*)p;
  v4f b = *(const v4f*)(p + 4);
  return __builtin_shufflevector(a, b, 0,1,2,3,4,5,6,7);
}

// ---------------------------------------------------------------------------
// Weight pre-pack: f32 -> f16 WMMA A-fragments (16x32, gfx1250 K-pattern).
// ---------------------------------------------------------------------------
__global__ void wavenet_pack(const float* __restrict__ conv_w,
                             const float* __restrict__ skip_w,
                             const float* __restrict__ out_w,
                             const float* __restrict__ out1_w,
                             _Float16* __restrict__ pack) {
  const int f    = blockIdx.x;
  const int lane = threadIdx.x;
  const int M    = lane & 15;
  const int hi   = lane >> 4;
  _Float16* dst = pack + ((size_t)f << 9) + (lane << 4);
#pragma unroll
  for (int h = 0; h < 16; ++h) {
    const int K = (h & 7) + ((h < 8) ? 0 : 16) + hi * 8;
    float v;
    if (f < NFRAG_CONV) {                 // dilated conv: (L,128,64,3)
      int L = f / 48, rem = f % 48;
      int m = rem / 6, tap = (rem % 6) / 2, kf = rem & 1;
      v = conv_w[(((size_t)L * 128 + (16 * m + M)) * 64 + (kf * 32 + K)) * 3 + tap];
    } else if (f < FRAG_OUT0) {           // skip 1x1: (L,64,64,1)
      int f2 = f - FRAG_SKIP0, L = f2 / 8, m = (f2 % 8) / 2, kf = f2 & 1;
      v = skip_w[((size_t)L * 64 + (16 * m + M)) * 64 + (kf * 32 + K)];
    } else if (f < FRAG_OUT10) {          // residual 1x1: (L,64,64,1)
      int f2 = f - FRAG_OUT0, L = f2 / 8, m = (f2 % 8) / 2, kf = f2 & 1;
      v = out_w[((size_t)L * 64 + (16 * m + M)) * 64 + (kf * 32 + K)];
    } else {                              // out1 1x1: (64,64,1)
      int f2 = f - FRAG_OUT10, m = f2 / 2, kf = f2 & 1;
      v = out1_w[(size_t)(16 * m + M) * 64 + (kf * 32 + K)];
    }
    dst[h] = (_Float16)v;
  }
}

// ---------------------------------------------------------------------------
// Fully fused WaveNet tile kernel. One WG = one (batch, 512-t tile).
// ---------------------------------------------------------------------------
__global__ __launch_bounds__(256, 2)
void wavenet_fused(const float* __restrict__ x,
                   const float* __restrict__ input_w, const float* __restrict__ input_b,
                   const float* __restrict__ conv_b,  const float* __restrict__ out_b,
                   const float* __restrict__ skip_b,  const float* __restrict__ out1_b,
                   const float* __restrict__ out2_w,  const float* __restrict__ out2_b,
                   const _Float16* __restrict__ pack,
                   float* __restrict__ out) {
  __shared__ __align__(16) _Float16 hbuf[HROWS * 64];     // residual stream, [row][ch]
  __shared__ __align__(16) _Float16 skipbuf[TILE * 64];   // skip accumulator, [t-t0][ch]
  __shared__ __align__(16) _Float16 gstage[8 * 16 * 64];  // per-wave g staging, [wave][j][ch]
  __shared__ __align__(16) float    biasf[256];  // [0..127]=conv_b [128..191]=skip_b [192..255]=out_b

  const int tid  = threadIdx.x;
  const int lane = tid & 31;
  const int wave = tid >> 5;
  const int jcol = lane & 15;           // N (time within 16-block)
  const int hi   = (lane >> 4) & 1;     // lane half
  const int b    = blockIdx.y;
  const int t0   = blockIdx.x * TILE;
  const int ringBase = t0 - HALO;

  // ---- input 1x1 conv + tanh fills the whole ring (zeros for t<0) ----
  {
    const int c  = tid & 63;
    const int r0 = tid >> 6;            // 0..3
    const float wc = input_w[c], bc = input_b[c];
    for (int row = r0; row < HROWS; row += 4) {
      const int t = ringBase + row;
      float hv = 0.f;
      if (t >= 0) hv = fast_tanh(wc * x[(size_t)b * TTOT + t] + bc);
      hbuf[row * 64 + c] = (_Float16)hv;
    }
    for (int i = tid; i < TILE * 64; i += 256) skipbuf[i] = (_Float16)0.f;
  }
  __syncthreads();

  // ---- 9 dilated gated layers ----
  for (int L = 0; L < NL; ++L) {
    const int d    = 1 << L;
    const int padL = (L < 5) ? 1024 : (1024 - (4 << L));   // ceil128 of remaining RF
    const int start = t0 - padL;
    const int nsb  = (TILE + padL) >> 7;

    if (tid < 128)      biasf[tid] = conv_b[L * 128 + tid];
    else if (tid < 192) biasf[tid] = skip_b[L * 64 + (tid - 128)];
    else                biasf[tid] = out_b[L * 64 + (tid - 192)];
    __syncthreads();

    for (int sb = nsb - 1; sb >= 0; --sb) {       // descending time order
      const int tb = start + sb * 128 + wave * 16;

      // ======== phase 1: all reads + all WMMAs ========
      v16h Bfrag[3][2];
#pragma unroll
      for (int tap = 0; tap < 3; ++tap) {
        const int row = tb + jcol - (2 - tap) * d - ringBase;
#pragma unroll
        for (int kf = 0; kf < 2; ++kf) Bfrag[tap][kf] = load_bfrag(hbuf, row, kf, hi);
      }

      // gated conv, one channel-tile pair (m / m+4) at a time; unroll-by-2
      // keeps A-load latency hidden while staying under 256 VGPRs.
      _Float16* st = gstage + wave * (16 * 64);
#pragma unroll 2
      for (int m = 0; m < 4; ++m) {
        v8f zt = load_biasC(biasf, m, hi);       // tanh half (rows 0..63)
        v8f zs = load_biasC(biasf, m + 4, hi);   // sigmoid half (rows 64..127)
#pragma unroll
        for (int tap = 0; tap < 3; ++tap)
#pragma unroll
          for (int kf = 0; kf < 2; ++kf) {
            v16h At = load_afrag(pack, ((L * 8 + m) * 3 + tap) * 2 + kf, lane);
            zt = WMMA_F16(At, Bfrag[tap][kf], zt);
            v16h As = load_afrag(pack, ((L * 8 + m + 4) * 3 + tap) * 2 + kf, lane);
            zs = WMMA_F16(As, Bfrag[tap][kf], zs);
          }
        union { u32x4 u; _Float16 h[8]; } pk;
#pragma unroll
        for (int r = 0; r < 8; ++r)
          pk.h[r] = (_Float16)(fast_tanh(zt[r]) * fast_sigmoid(zs[r]));
        *(u32x4*)(st + jcol * 64 + m * 16 + hi * 8) = pk.u;   // one b128 store
      }
      asm volatile("s_wait_dscnt 0" ::: "memory");   // staging store -> load, same wave

      v16h Gf[2];
#pragma unroll
      for (int kf = 0; kf < 2; ++kf) Gf[kf] = load_bfrag(st, jcol, kf, hi);

      v8f sk[4], hd[4];
#pragma unroll
      for (int m = 0; m < 4; ++m) {
        v8f a1 = load_biasC(biasf + 128, m, hi);   // skip bias
        v8f a2 = load_biasC(biasf + 192, m, hi);   // residual bias
#pragma unroll
        for (int kf = 0; kf < 2; ++kf) {
          v16h As = load_afrag(pack, FRAG_SKIP0 + ((L * 4 + m) * 2 + kf), lane);
          a1 = WMMA_F16(As, Gf[kf], a1);
          v16h Ao = load_afrag(pack, FRAG_OUT0 + ((L * 4 + m) * 2 + kf), lane);
          a2 = WMMA_F16(Ao, Gf[kf], a2);
        }
        sk[m] = a1; hd[m] = a2;
      }

      __syncthreads();   // all tap reads done before anyone writes this super-block

      // ======== phase 2: own-block writes only (b128 RMW per tile) ========
      if (tb >= 0) {
        const int row = tb + jcol - ringBase;
#pragma unroll
        for (int m = 0; m < 4; ++m) {
          u32x4* hp = (u32x4*)(hbuf + row * 64 + m * 16 + hi * 8);
          union { u32x4 u; _Float16 h[8]; } pk;
          pk.u = *hp;
#pragma unroll
          for (int r = 0; r < 8; ++r)
            pk.h[r] = (_Float16)((float)pk.h[r] + hd[m][r]);
          *hp = pk.u;
        }
        if (tb >= t0) {
          const int srow = tb + jcol - t0;
#pragma unroll
          for (int m = 0; m < 4; ++m) {
            u32x4* sp = (u32x4*)(skipbuf + srow * 64 + m * 16 + hi * 8);
            union { u32x4 u; _Float16 h[8]; } pk;
            pk.u = *sp;
#pragma unroll
            for (int r = 0; r < 8; ++r)
              pk.h[r] = (_Float16)((float)pk.h[r] + sk[m][r]);
            *sp = pk.u;
          }
        }
      }
      // no trailing barrier: next (lower) super-block reads strictly below
      // this one's write region.
    }
    __syncthreads();   // layer complete before bias restage / next-layer reads
  }

  // ---- head: y = tanh(out1 @ skip + b1);  out = out2 @ y + b2 ----
  if (tid < 64) biasf[tid] = out1_b[tid];
  __syncthreads();
  const float ob2 = out2_b[0];
#pragma unroll
  for (int ii = 0; ii < TILE / 128; ++ii) {
    const int blk = wave + 8 * ii;            // 0..31
    const int tb  = t0 + blk * 16;
    v16h Sf[2];
#pragma unroll
    for (int kf = 0; kf < 2; ++kf) Sf[kf] = load_bfrag(skipbuf, blk * 16 + jcol, kf, hi);
    float partial = 0.f;
#pragma unroll
    for (int m = 0; m < 4; ++m) {
      v8f y = load_biasC(biasf, m, hi);
#pragma unroll
      for (int kf = 0; kf < 2; ++kf) {
        v16h A1 = load_afrag(pack, FRAG_OUT10 + m * 2 + kf, lane);
        y = WMMA_F16(A1, Sf[kf], y);
      }
      v8f w2 = load_biasC(out2_w, m, hi);
#pragma unroll
      for (int r = 0; r < 8; ++r) partial += w2[r] * fast_tanh(y[r]);
    }
    const float total = partial + __shfl_xor(partial, 16, 32);
    if (lane < 16) out[(size_t)b * TTOT + tb + jcol] = total + ob2;
  }
}

// ---------------------------------------------------------------------------
extern "C" void kernel_launch(void* const* d_in, const int* in_sizes, int n_in,
                              void* d_out, int out_size, void* d_ws, size_t ws_size,
                              hipStream_t stream) {
  (void)in_sizes; (void)n_in; (void)out_size; (void)ws_size;
  const float* x        = (const float*)d_in[0];
  const float* input_w  = (const float*)d_in[1];
  const float* input_b  = (const float*)d_in[2];
  const float* conv_w   = (const float*)d_in[3];
  const float* conv_b   = (const float*)d_in[4];
  const float* out_w    = (const float*)d_in[5];
  const float* out_b    = (const float*)d_in[6];
  const float* skip_w   = (const float*)d_in[7];
  const float* skip_b   = (const float*)d_in[8];
  const float* out1_w   = (const float*)d_in[9];
  const float* out1_b   = (const float*)d_in[10];
  const float* out2_w   = (const float*)d_in[11];
  const float* out2_b   = (const float*)d_in[12];

  _Float16* pack = (_Float16*)d_ws;   // 584 fragments * 1KB ≈ 0.6 MB

  wavenet_pack<<<dim3(NFRAGS), dim3(32), 0, stream>>>(conv_w, skip_w, out_w, out1_w, pack);

  dim3 grid(TTOT / TILE, NB);         // 64 x 8 workgroups
  wavenet_fused<<<grid, dim3(256), 0, stream>>>(
      x, input_w, input_b, conv_b, out_b, skip_b, out1_b, out2_w, out2_b,
      pack, (float*)d_out);
}
